// Head_9414568313057
// MI455X (gfx1250) — compile-verified, matrix operand order checked by reading, not run on hardware
//
#include <hip/hip_runtime.h>
#include <hip/hip_bf16.h>
#include <cstdint>

// ---------------------------------------------------------------------------
// Types for CDNA5 WMMA (wave32): bf16 16x16x32, f32 accumulate.
// ---------------------------------------------------------------------------
typedef __attribute__((ext_vector_type(16))) __bf16          v16bf;
typedef __attribute__((ext_vector_type(8)))  __bf16          v8bf;
typedef __attribute__((ext_vector_type(8)))  float           v8f;
typedef __attribute__((ext_vector_type(8)))  float           f8;
typedef __attribute__((ext_vector_type(8)))  unsigned short  us8;

#define WMMA_BF16(a, b, c) \
  __builtin_amdgcn_wmma_f32_16x16x32_bf16(false, (a), false, (b), (short)0, (c), false, false)

static __device__ __forceinline__ unsigned short f2bfu(float f) {
  unsigned u = __builtin_bit_cast(unsigned, f);
  return (unsigned short)(u >> 16);
}
// Fragment builders: pure register concat (no per-half moves).
static __device__ __forceinline__ v16bf mk16(us8 g0, us8 g1) {
  v8bf a = __builtin_bit_cast(v8bf, g0);
  v8bf b = __builtin_bit_cast(v8bf, g1);
  return __builtin_shufflevector(a, b, 0,1,2,3,4,5,6,7,8,9,10,11,12,13,14,15);
}
static __device__ __forceinline__ v16bf mk16f(f8 g0, f8 g1) {
  v8bf a = __builtin_convertvector(g0, v8bf);   // packed v_cvt f32->bf16
  v8bf b = __builtin_convertvector(g1, v8bf);
  return __builtin_shufflevector(a, b, 0,1,2,3,4,5,6,7,8,9,10,11,12,13,14,15);
}

// Problem constants (fixed by the reference).
#define BATCH 4
#define SEQ   4096
#define CDIM  1024
#define HDIM  64
#define ROWS  (BATCH * SEQ)     // 16384
#define SCALE 0.25f             // HEAD_COUNT(16) ** -0.5

// ---------------------------------------------------------------------------
// Kernel 1: transpose+quantize weights:  W[C,H] f32  ->  Wt[3][H][C] bf16
// ---------------------------------------------------------------------------
__global__ void wprep_kernel(const float* __restrict__ Wq,
                             const float* __restrict__ Wk,
                             const float* __restrict__ Wv,
                             unsigned short* __restrict__ wt) {
  int idx = blockIdx.x * 256 + threadIdx.x;
  const int per = HDIM * CDIM;               // 65536
  if (idx < 3 * per) {
    int m   = idx / per;
    int rem = idx - m * per;
    int n   = rem >> 10;
    int k   = rem & 1023;
    const float* W = (m == 0) ? Wq : ((m == 1) ? Wk : Wv);
    wt[idx] = f2bfu(W[(size_t)k * HDIM + n]);
  }
}

// ---------------------------------------------------------------------------
// Kernel 2: fused QKV projection, software-pipelined WMMA.
// ---------------------------------------------------------------------------
__global__ void __launch_bounds__(256) qkv_kernel(
    const float* __restrict__ x, const unsigned short* __restrict__ wt,
    unsigned short* __restrict__ qb, unsigned short* __restrict__ kb,
    unsigned short* __restrict__ vt) {
  const int wave = threadIdx.x >> 5;
  const int lane = threadIdx.x & 31;
  const int ln   = lane & 15;
  const int hl   = lane >> 4;
  const int r0   = (blockIdx.x * 8 + wave) * 16;

  const float*          xb = x  + (size_t)(r0 + ln) * CDIM + hl * 8;
  const unsigned short* wb = wt + (size_t)ln * CDIM + hl * 8;

  auto loadA = [&](int kc) {
    const float* p = xb + kc * 32;
    return mk16f(*(const f8*)p, *(const f8*)(p + 16));
  };
  auto loadB = [&](int kc, int i) {
    int m = i >> 2, n = i & 3;
    const unsigned short* p = wb + m * (HDIM * CDIM) + n * 16 * CDIM + kc * 32;
    return mk16(*(const us8*)p, *(const us8*)(p + 16));
  };

  v8f acc[12];
#pragma unroll
  for (int i = 0; i < 12; ++i) acc[i] = (v8f)0.0f;

  // Pipeline: A one chunk ahead, B one fragment ahead.
  v16bf a_cur = loadA(0);
  v16bf b_cur = loadB(0, 0);
  for (int kc = 0; kc < CDIM / 32; ++kc) {
    v16bf a_next = a_cur;
    if (kc + 1 < CDIM / 32) a_next = loadA(kc + 1);
#pragma unroll
    for (int i = 0; i < 12; ++i) {
      v16bf b_next = b_cur;
      if (i + 1 < 12)               b_next = loadB(kc, i + 1);
      else if (kc + 1 < CDIM / 32)  b_next = loadB(kc + 1, 0);
      acc[i] = WMMA_BF16(a_cur, b_cur, acc[i]);
      b_cur = b_next;
    }
    a_cur = a_next;
  }

#pragma unroll
  for (int n = 0; n < 4; ++n) {
#pragma unroll
    for (int j = 0; j < 8; ++j) {
      int row = r0 + j + hl * 8;
      int col = n * 16 + ln;
      qb[(size_t)row * HDIM + col] = f2bfu(acc[0 * 4 + n][j]);
      kb[(size_t)row * HDIM + col] = f2bfu(acc[1 * 4 + n][j]);
      int b = row >> 12;
      int t = row & 4095;
      vt[((size_t)b * HDIM + col) * SEQ + t] = f2bfu(acc[2 * 4 + n][j]);  // V^T
    }
  }
}

// ---------------------------------------------------------------------------
// Kernel 3: causal flash attention.
// K tiles double-buffered in LDS via GLOBAL_LOAD_ASYNC_TO_LDS_B128 (ASYNCcnt),
// overlapped with softmax + WMMA. P-transpose scratch reuses the dead K buffer.
// ---------------------------------------------------------------------------
__global__ void __launch_bounds__(256) attn_kernel(
    const unsigned short* __restrict__ qb, const unsigned short* __restrict__ kb,
    const unsigned short* __restrict__ vt, float* __restrict__ out) {
  const int wave = threadIdx.x >> 5;
  const int lane = threadIdx.x & 31;
  const int ln   = lane & 15;
  const int hl   = lane >> 4;

  const int g  = blockIdx.x * 8 + wave;
  const int b  = g >> 8;
  const int t0 = (g & 255) * 16;

  const unsigned short* qB = qb + (size_t)b * SEQ * HDIM;
  const unsigned short* kB = kb + (size_t)b * SEQ * HDIM;
  const unsigned short* vB = vt + (size_t)b * HDIM * SEQ;

  // Per-wave double-buffered K tile: 2 x (32 keys x 64 dims) bf16 = 2 x 4KB.
  __shared__ unsigned short kbuf[8][2][32 * HDIM];   // 64 KB total

  // Issue an async 4KB tile copy: 8 x b128 per lane, tracked by ASYNCcnt.
  auto issueK = [&](unsigned ldsbase, const unsigned short* src) {
    unsigned long long ga = (unsigned long long)(uintptr_t)src;
#pragma unroll
    for (int i = 0; i < 8; ++i) {
      unsigned           lo = ldsbase + (unsigned)((i * 32 + lane) * 16);
      unsigned long long gp = ga + (unsigned long long)((i * 32 + lane) * 16);
      asm volatile("global_load_async_to_lds_b128 %0, %1, off"
                   :: "v"(lo), "v"(gp) : "memory");
    }
  };

  // Q A-fragments (loaded once).
  v16bf qa[2];
#pragma unroll
  for (int c = 0; c < 2; ++c) {
    const unsigned short* qp = qB + (size_t)(t0 + ln) * HDIM + c * 32 + hl * 8;
    qa[c] = mk16(*(const us8*)qp, *(const us8*)(qp + 16));
  }

  float mrow[8], lrow[8], al[8];
#pragma unroll
  for (int j = 0; j < 8; ++j) { mrow[j] = -__builtin_inff(); lrow[j] = 0.0f; }
  v8f o[4];
#pragma unroll
  for (int n = 0; n < 4; ++n) o[n] = (v8f)0.0f;

  unsigned kbase[2] = { (unsigned)(uintptr_t)&kbuf[wave][0][0],
                        (unsigned)(uintptr_t)&kbuf[wave][1][0] };
  issueK(kbase[0], kB);                     // prologue: tile j0 = 0
  int cur = 0;

  for (int j0 = 0; j0 < t0 + 16; j0 += 32) {
    const bool hasNext = (j0 + 32 < t0 + 16);
    if (hasNext) {
      issueK(kbase[cur ^ 1], kB + (size_t)(j0 + 32) * HDIM);
      __builtin_prefetch(vB + (size_t)ln * SEQ + j0 + 32, 0, 1);  // global_prefetch_b8
      asm volatile("s_wait_asynccnt 0x8" ::: "memory");   // current tile landed
    } else {
      asm volatile("s_wait_asynccnt 0x0" ::: "memory");
    }
    unsigned short* kt = &kbuf[wave][cur][0];

    // ---- preload V B-fragments (independent of softmax) ------------------
    v16bf vb[4];
#pragma unroll
    for (int n = 0; n < 4; ++n) {
      const unsigned short* vp = vB + (size_t)(n * 16 + ln) * SEQ + j0 + hl * 8;
      vb[n] = mk16(*(const us8*)vp, *(const us8*)(vp + 16));
    }

    // ---- S = Q K^T from the LDS tile -------------------------------------
    v8f sc0 = (v8f)0.0f, sc1 = (v8f)0.0f;
#pragma unroll
    for (int c = 0; c < 2; ++c) {
      const unsigned short* kp0 = kt + (size_t)ln * HDIM + c * 32 + hl * 8;
      const unsigned short* kp1 = kt + (size_t)(16 + ln) * HDIM + c * 32 + hl * 8;
      v16bf b0 = mk16(*(const us8*)kp0, *(const us8*)(kp0 + 16));
      v16bf b1 = mk16(*(const us8*)kp1, *(const us8*)(kp1 + 16));
      sc0 = WMMA_BF16(qa[c], b0, sc0);
      sc1 = WMMA_BF16(qa[c], b1, sc1);
    }

    // ---- scale + causal mask ---------------------------------------------
    const bool needMask = (j0 + 31 > t0);
#pragma unroll
    for (int j = 0; j < 8; ++j) {
      float s0 = sc0[j] * SCALE;
      float s1 = sc1[j] * SCALE;
      if (needMask) {
        int row = t0 + j + hl * 8;
        int k0  = j0 + ln;
        s0 = (k0 > row)      ? -__builtin_inff() : s0;
        s1 = (k0 + 16 > row) ? -__builtin_inff() : s1;
      }
      sc0[j] = s0; sc1[j] = s1;
    }

    // ---- online softmax; P tile written into the (now dead) K buffer -----
#pragma unroll
    for (int j = 0; j < 8; ++j) {
      float v = fmaxf(sc0[j], sc1[j]);
#pragma unroll
      for (int m = 8; m >= 1; m >>= 1) v = fmaxf(v, __shfl_xor(v, m));
      float mn    = fmaxf(mrow[j], v);
      float alpha = __expf(mrow[j] - mn);
      float p0    = __expf(sc0[j] - mn);
      float p1    = __expf(sc1[j] - mn);
      float rs    = p0 + p1;
#pragma unroll
      for (int m = 8; m >= 1; m >>= 1) rs += __shfl_xor(rs, m);
      lrow[j] = lrow[j] * alpha + rs;
      mrow[j] = mn;
      al[j]   = alpha;
      int rl = j + hl * 8;
      kt[rl * 32 + ln]      = f2bfu(p0);    // C/D layout -> LDS
      kt[rl * 32 + 16 + ln] = f2bfu(p1);
    }
#pragma unroll
    for (int n = 0; n < 4; ++n)
#pragma unroll
      for (int j = 0; j < 8; ++j) o[n][j] *= al[j];

    // ---- reload P in A layout, then O += P V -----------------------------
    const unsigned short* pp = kt + ln * 32 + hl * 8;
    v16bf pa = mk16(*(const us8*)pp, *(const us8*)(pp + 16));
#pragma unroll
    for (int n = 0; n < 4; ++n) o[n] = WMMA_BF16(pa, vb[n], o[n]);

    cur ^= 1;
  }

  // ---- finalize ----------------------------------------------------------
#pragma unroll
  for (int n = 0; n < 4; ++n) {
#pragma unroll
    for (int j = 0; j < 8; ++j) {
      int row = t0 + j + hl * 8;
      out[((size_t)b * SEQ + row) * HDIM + n * 16 + ln] = o[n][j] * (1.0f / lrow[j]);
    }
  }
}

// ---------------------------------------------------------------------------
// Launcher
// ---------------------------------------------------------------------------
extern "C" void kernel_launch(void* const* d_in, const int* in_sizes, int n_in,
                              void* d_out, int out_size, void* d_ws, size_t ws_size,
                              hipStream_t stream) {
  (void)in_sizes; (void)n_in; (void)out_size; (void)ws_size;
  const float* x  = (const float*)d_in[0];
  const float* Wq = (const float*)d_in[1];
  const float* Wk = (const float*)d_in[2];
  const float* Wv = (const float*)d_in[3];
  float* out = (float*)d_out;

  char* ws = (char*)d_ws;
  unsigned short* wt = (unsigned short*)(ws);                          // 384 KB
  unsigned short* qb = (unsigned short*)(ws + (size_t)393216);         // 2 MB
  unsigned short* kb = (unsigned short*)(ws + (size_t)393216 + 2097152);
  unsigned short* vt = (unsigned short*)(ws + (size_t)393216 + 2 * 2097152);

  wprep_kernel<<<(3 * HDIM * CDIM + 255) / 256, 256, 0, stream>>>(Wq, Wk, Wv, wt);
  qkv_kernel<<<ROWS / 128, 256, 0, stream>>>(x, wt, qb, kb, vt);
  attn_kernel<<<ROWS / 128, 256, 0, stream>>>(qb, kb, vt, out);
}
